// Transformer_4647154614738
// MI455X (gfx1250) — matrix operand order repaired and checked by reading, not compiled, not run on hardware
//
#include <hip/hip_runtime.h>
#include <hip/hip_bf16.h>

typedef __attribute__((ext_vector_type(16))) _Float16 v16h;
typedef __attribute__((ext_vector_type(8)))  _Float16 v8h;
typedef __attribute__((ext_vector_type(4)))  _Float16 h4;
typedef __attribute__((ext_vector_type(8)))  float    v8f;
typedef __attribute__((ext_vector_type(4)))  float    v4f;

#define DV   1024
#define FFV  2048
#define HV   8
#define DH   128
#define NL   6
#define BV   8
#define NQ   1024
#define MROWS (BV * NQ)          // 8192
#define KSTEP 32

__device__ __forceinline__ v8f wmma16(v16h a, v16h b, v8f c) {
  return __builtin_amdgcn_wmma_f32_16x16x32_f16(false, a, false, b, (short)0, c,
                                                false, false);
}

__device__ __forceinline__ v8h pack8(v4f a, v4f b) {
  h4 ha = __builtin_convertvector(a, h4);
  h4 hb = __builtin_convertvector(b, h4);
  return __builtin_shufflevector(ha, hb, 0, 1, 2, 3, 4, 5, 6, 7);
}
// load 8 contiguous f32, convert to packed f16
__device__ __forceinline__ v8h cvt8(const float* p) {
  return pack8(*(const v4f*)p, *(const v4f*)(p + 4));
}
__device__ __forceinline__ v8h ld8h(const _Float16* p) { return *(const v8h*)p; }
__device__ __forceinline__ v16h cat16(v8h lo, v8h hi) {
  return __builtin_shufflevector(lo, hi, 0, 1, 2, 3, 4, 5, 6, 7, 8, 9, 10, 11,
                                 12, 13, 14, 15);
}

// ---------------------------------------------------------------------------
// Generic GEMM: C[M,N] = act(A[M,K] @ W[N,K]^T + bias[N])
// 256 threads (8 waves, 4x2), block tile 128(M) x 64(N), K step 32.
// Software pipeline: double-buffered LDS; global loads for tile k+1 issued
// before the WMMAs of tile k, converted/stored after. One barrier per step.
// ---------------------------------------------------------------------------
__global__ __launch_bounds__(256)
void gemm_wmma(const float* __restrict__ A, const float* __restrict__ W,
               const float* __restrict__ bias, float* __restrict__ C,
               int M, int N, int K, int ldc, int relu) {
  __shared__ alignas(16) _Float16 As[2][128][KSTEP];  // 16 KB
  __shared__ alignas(16) _Float16 Wt[2][KSTEP][64];   //  8 KB (k-major)

  const int tid  = threadIdx.x;
  const int lane = tid & 31;
  const int wid  = tid >> 5;
  const int wm   = wid >> 1;       // 0..3  (32-row strip)
  const int wn   = wid & 1;        // 0..1  (32-col strip)
  const int m0   = blockIdx.x * 128;
  const int n0   = blockIdx.y * 64;

  const int mrow  = lane & 15;
  const int khalf = lane >> 4;

  // per-thread staging coordinates
  const int ar0 = (tid + 0)   >> 2, ac0 = (tid & 3) * 8;          // A chunk 0
  const int ar1 = (tid + 256) >> 2, ac1 = ac0;                    // A chunk 1
  const int wr0 = (tid + 0)   >> 3, wc0 = (tid & 7) * 4;          // W chunk 0
  const int wr1 = (tid + 256) >> 3, wc1 = wc0;                    // W chunk 1

  v4f ra[2][2], rw[2];
  auto g_load = [&](int k0) {
    const float* p0 = A + (size_t)(m0 + ar0) * K + k0 + ac0;
    const float* p1 = A + (size_t)(m0 + ar1) * K + k0 + ac1;
    ra[0][0] = *(const v4f*)p0;       ra[0][1] = *(const v4f*)(p0 + 4);
    ra[1][0] = *(const v4f*)p1;       ra[1][1] = *(const v4f*)(p1 + 4);
    rw[0] = *(const v4f*)(W + (size_t)(n0 + wr0) * K + k0 + wc0);
    rw[1] = *(const v4f*)(W + (size_t)(n0 + wr1) * K + k0 + wc1);
  };
  auto l_store = [&](int buf) {
    *(v8h*)&As[buf][ar0][ac0] = pack8(ra[0][0], ra[0][1]);
    *(v8h*)&As[buf][ar1][ac1] = pack8(ra[1][0], ra[1][1]);
    h4 h0 = __builtin_convertvector(rw[0], h4);
    h4 h1 = __builtin_convertvector(rw[1], h4);
#pragma unroll
    for (int j = 0; j < 4; ++j) {
      Wt[buf][wc0 + j][wr0] = h0[j];
      Wt[buf][wc1 + j][wr1] = h1[j];
    }
  };

  v8f acc[2][2] = {};

  g_load(0);
  l_store(0);
  __syncthreads();

  int buf = 0;
  for (int k0 = 0; k0 < K; k0 += KSTEP) {
    const bool more = (k0 + KSTEP) < K;
    if (more) {
      g_load(k0 + KSTEP);                       // issue loads, no wait yet
      if (k0 + 2 * KSTEP < K)                   // L2 prefetch 2 tiles ahead
        __builtin_prefetch(A + (size_t)(m0 + (tid >> 1)) * K + k0 + 2 * KSTEP +
                               (tid & 1) * 16, 0, 0);
    }

    v16h a[2], b[2];
#pragma unroll
    for (int mi = 0; mi < 2; ++mi) {
      const _Float16* ar = &As[buf][wm * 32 + mi * 16 + mrow][khalf * 8];
      a[mi] = cat16(ld8h(ar), ld8h(ar + 16));
    }
#pragma unroll
    for (int ni = 0; ni < 2; ++ni) {
      const int nc = wn * 32 + ni * 16 + mrow;
#pragma unroll
      for (int e = 0; e < 16; ++e)
        b[ni][e] = Wt[buf][khalf * 8 + e + (e >= 8 ? 8 : 0)][nc];
    }
#pragma unroll
    for (int mi = 0; mi < 2; ++mi)
#pragma unroll
      for (int ni = 0; ni < 2; ++ni)
        acc[mi][ni] = wmma16(a[mi], b[ni], acc[mi][ni]);

    if (more) l_store(buf ^ 1);                 // waits happen here, post-WMMA
    __syncthreads();
    buf ^= 1;
  }

  const int nn = lane & 15;
#pragma unroll
  for (int mi = 0; mi < 2; ++mi)
#pragma unroll
    for (int ni = 0; ni < 2; ++ni) {
      int col = n0 + wn * 32 + ni * 16 + nn;
      float bv = bias[col];
#pragma unroll
      for (int r = 0; r < 8; ++r) {
        int row = m0 + wm * 32 + mi * 16 + r + 8 * khalf;
        float v = acc[mi][ni][r] + bv;
        if (relu) v = fmaxf(v, 0.0f);
        C[(size_t)row * ldc + col] = v;
      }
    }
}

// ---------------------------------------------------------------------------
// Attention: one wave per (batch, head, 16-query tile).
// qkv rows: (b*1024 + t), row stride 3*D, cols [q | k | v].
// Pass 1 computes S^T = K @ Q^T so the streamed K operand is the A-side
// (contiguous per-lane 32B runs) and the one-time Q operand is the B-side.
// Scores 16x1024 fp32 in LDS, fp32 softmax, P@V via WMMA.
// ---------------------------------------------------------------------------
__global__ __launch_bounds__(32)
void attn_wmma(const float* __restrict__ qkv, float* __restrict__ out, int causal) {
  __shared__ float S[16][NQ];    // 64 KB

  const int lane  = threadIdx.x;
  const int q0    = blockIdx.x * 16;
  const int head  = blockIdx.y;
  const int batch = blockIdx.z;

  const size_t rs = 3 * DV;
  const float* Qp = qkv + (size_t)batch * NQ * rs + head * DH;
  const float* Kp = Qp + DV;
  const float* Vp = Qp + 2 * DV;

  const int mrow  = lane & 15;
  const int khalf = lane >> 4;
  const float scale = 0.08838834764831845f;   // 1/sqrt(128)

  // one-time Q preload as B-fragments (lane&15 = query, element = dh offset)
  v16h qb[4];
#pragma unroll
  for (int kc = 0; kc < 4; ++kc)
#pragma unroll
    for (int e = 0; e < 16; ++e)
      qb[kc][e] = (_Float16)Qp[(size_t)(q0 + mrow) * rs + kc * 32 + khalf * 8 + e + (e >= 8 ? 8 : 0)];

  // S^T = K Q^T * scale (+ causal mask): A = K tile rows (vectorized)
  for (int kt = 0; kt < NQ / 16; ++kt) {
    v8f s = {};
#pragma unroll
    for (int kc = 0; kc < 4; ++kc) {
      const float* kr = Kp + (size_t)(kt * 16 + mrow) * rs + kc * 32 + khalf * 8;
      v16h kf = cat16(cvt8(kr), cvt8(kr + 16));
      s = wmma16(kf, qb[kc], s);
    }
    // C: lane&15 = query, vgpr r -> key = kt*16 + r + 8*khalf (contiguous)
    const int q  = lane & 15;
    const int kb = kt * 16 + 8 * khalf;
    v4f lo, hi;
#pragma unroll
    for (int j = 0; j < 4; ++j) {
      float v0 = s[j] * scale;
      float v1 = s[j + 4] * scale;
      if (causal && (kb + j     > q0 + q)) v0 = -1e30f;
      if (causal && (kb + j + 4 > q0 + q)) v1 = -1e30f;
      lo[j] = v0; hi[j] = v1;
    }
    *(v4f*)&S[q][kb]     = lo;
    *(v4f*)&S[q][kb + 4] = hi;
  }
  __syncthreads();

  // fp32 softmax: lane L handles row L&15, half L>>4 (512 keys), shfl combine
  {
    const int row = lane & 15, hh = lane >> 4;
    const int j0 = hh * 512;
    float mx = -3.4e38f;
    for (int i = 0; i < 128; ++i) {
      v4f xv = *(const v4f*)&S[row][j0 + 4 * i];
      mx = fmaxf(mx, fmaxf(fmaxf(xv[0], xv[1]), fmaxf(xv[2], xv[3])));
    }
    mx = fmaxf(mx, __shfl_xor(mx, 16));
    float sum = 0.0f;
    for (int i = 0; i < 128; ++i) {
      v4f xv = *(const v4f*)&S[row][j0 + 4 * i];
      xv[0] = __expf(xv[0] - mx);
      xv[1] = __expf(xv[1] - mx);
      xv[2] = __expf(xv[2] - mx);
      xv[3] = __expf(xv[3] - mx);
      *(v4f*)&S[row][j0 + 4 * i] = xv;
      sum += xv[0] + xv[1] + xv[2] + xv[3];
    }
    sum += __shfl_xor(sum, 16);
    float inv = 1.0f / sum;
    for (int i = 0; i < 128; ++i) {
      v4f xv = *(const v4f*)&S[row][j0 + 4 * i];
      xv *= inv;
      *(v4f*)&S[row][j0 + 4 * i] = xv;
    }
  }
  __syncthreads();

  // O = P @ V   (16 x 128), keys in steps of 32; P vectorized, V gathered per element
  v8f o[8] = {};
  for (int kt = 0; kt < NQ / 32; ++kt) {
    const float* sr = &S[mrow][kt * 32 + khalf * 8];
    v16h a = cat16(cvt8(sr), cvt8(sr + 16));
#pragma unroll
    for (int nt = 0; nt < 8; ++nt) {
      v16h bf;   // element = key, lane&15 = dh
#pragma unroll
      for (int e = 0; e < 16; ++e)
        bf[e] = (_Float16)Vp[(size_t)(kt * 32 + khalf * 8 + e + (e >= 8 ? 8 : 0)) * rs + nt * 16 + mrow];
      o[nt] = wmma16(a, bf, o[nt]);
    }
  }

#pragma unroll
  for (int nt = 0; nt < 8; ++nt)
#pragma unroll
    for (int r = 0; r < 8; ++r) {
      int qrow = q0 + r + 8 * khalf;
      int col  = head * DH + nt * 16 + (lane & 15);
      out[((size_t)batch * NQ + qrow) * DV + col] = o[nt][r];
    }
}

// ---------------------------------------------------------------------------
// h = LayerNorm(h + r) * g + b   (one 256-thread block per row, D=1024)
// ---------------------------------------------------------------------------
__global__ __launch_bounds__(256)
void add_ln(float* __restrict__ h, const float* __restrict__ r,
            const float* __restrict__ g, const float* __restrict__ bta) {
  __shared__ float red[8];
  const int tid = threadIdx.x;
  const int lane = tid & 31, wid = tid >> 5;
  const size_t base = (size_t)blockIdx.x * DV;
  const int c = tid * 4;

  v4f v = *(const v4f*)(h + base + c) + *(const v4f*)(r + base + c);
  float s = v[0] + v[1] + v[2] + v[3];
#pragma unroll
  for (int off = 16; off > 0; off >>= 1) s += __shfl_xor(s, off);
  if (lane == 0) red[wid] = s;
  __syncthreads();
  float tot = 0.0f;
#pragma unroll
  for (int i = 0; i < 8; ++i) tot += red[i];
  const float mu = tot * (1.0f / DV);
  __syncthreads();

  v4f d = v - mu;
  float s2 = d[0] * d[0] + d[1] * d[1] + d[2] * d[2] + d[3] * d[3];
#pragma unroll
  for (int off = 16; off > 0; off >>= 1) s2 += __shfl_xor(s2, off);
  if (lane == 0) red[wid] = s2;
  __syncthreads();
  float tot2 = 0.0f;
#pragma unroll
  for (int i = 0; i < 8; ++i) tot2 += red[i];
  const float inv = rsqrtf(tot2 * (1.0f / DV) + 1e-5f);

  v4f g4 = *(const v4f*)(g + c);
  v4f b4 = *(const v4f*)(bta + c);
  v4f outv = d * inv * g4 + b4;
  *(v4f*)(h + base + c) = outv;
}

// ---------------------------------------------------------------------------
// Build initial target sequence h[b, t, :].
// ---------------------------------------------------------------------------
__global__ __launch_bounds__(256)
void build_h(const float* __restrict__ sq, const float* __restrict__ one_e,
             const float* __restrict__ zero_e, const float* __restrict__ label,
             float* __restrict__ h) {
  const int row = blockIdx.x;
  const int t = row & (NQ - 1);
  const float* src = (t == 0) ? sq : ((label[t - 1] > 0.5f) ? one_e : zero_e);
  const int c = threadIdx.x * 4;
  *(v4f*)&h[(size_t)row * DV + c] = *(const v4f*)&src[c];
}

// ---------------------------------------------------------------------------
// out[row] = sigmoid(h[row,:] . w + b)   (one wave per row)
// ---------------------------------------------------------------------------
__global__ __launch_bounds__(256)
void cls_head(const float* __restrict__ h, const float* __restrict__ w,
              const float* __restrict__ b, float* __restrict__ out) {
  const int tid = threadIdx.x;
  const int lane = tid & 31, wid = tid >> 5;
  const int row = blockIdx.x * 8 + wid;
  const float* hr = h + (size_t)row * DV + lane * 32;
  const float* wr = w + lane * 32;
  float s = 0.0f;
#pragma unroll
  for (int i = 0; i < 8; ++i) {
    v4f a = *(const v4f*)(hr + i * 4);
    v4f c = *(const v4f*)(wr + i * 4);
    s += a[0] * c[0] + a[1] * c[1] + a[2] * c[2] + a[3] * c[3];
  }
#pragma unroll
  for (int off = 16; off > 0; off >>= 1) s += __shfl_xor(s, off);
  if (lane == 0) out[row] = 1.0f / (1.0f + __expf(-(s + b[0])));
}

// ---------------------------------------------------------------------------
extern "C" void kernel_launch(void* const* d_in, const int* in_sizes, int n_in,
                              void* d_out, int out_size, void* d_ws, size_t ws_size,
                              hipStream_t stream) {
  const float* x        = (const float*)d_in[0];
  const float* label    = (const float*)d_in[1];
  const float* start_q  = (const float*)d_in[2];
  const float* one_emb  = (const float*)d_in[3];
  const float* zero_emb = (const float*)d_in[4];
  const float* sa_in_w  = (const float*)d_in[5];
  const float* sa_in_b  = (const float*)d_in[6];
  const float* sa_out_w = (const float*)d_in[7];
  const float* sa_out_b = (const float*)d_in[8];
  const float* ca_in_w  = (const float*)d_in[9];
  const float* ca_in_b  = (const float*)d_in[10];
  const float* ca_out_w = (const float*)d_in[11];
  const float* ca_out_b = (const float*)d_in[12];
  const float* lin1_w   = (const float*)d_in[13];
  const float* lin1_b   = (const float*)d_in[14];
  const float* lin2_w   = (const float*)d_in[15];
  const float* lin2_b   = (const float*)d_in[16];
  const float* ln_g     = (const float*)d_in[17];
  const float* ln_b     = (const float*)d_in[18];
  const float* cls_w    = (const float*)d_in[19];
  const float* cls_b    = (const float*)d_in[20];
  float* out = (float*)d_out;

  float* ws = (float*)d_ws;
  float* h  = ws;                               // M*D
  float* t0 = h  + (size_t)MROWS * DV;          // M*3D  (qkv / q|kv)
  float* t1 = t0 + (size_t)MROWS * 3 * DV;      // M*D   (attn out / ffn out)
  float* t2 = t1 + (size_t)MROWS * DV;          // M*FF  (proj out / ffn mid)

  const dim3 blk256(256), blk32(32);
  const dim3 attn_grid(NQ / 16, HV, BV);

  build_h<<<MROWS, blk256, 0, stream>>>(start_q, one_emb, zero_emb, label, h);

  for (int l = 0; l < NL; ++l) {
    const float* saw  = sa_in_w  + (size_t)l * 3 * DV * DV;
    const float* sab  = sa_in_b  + (size_t)l * 3 * DV;
    const float* saow = sa_out_w + (size_t)l * DV * DV;
    const float* saob = sa_out_b + (size_t)l * DV;
    const float* caw  = ca_in_w  + (size_t)l * 3 * DV * DV;
    const float* cab  = ca_in_b  + (size_t)l * 3 * DV;
    const float* caow = ca_out_w + (size_t)l * DV * DV;
    const float* caob = ca_out_b + (size_t)l * DV;
    const float* l1w  = lin1_w   + (size_t)l * FFV * DV;
    const float* l1b  = lin1_b   + (size_t)l * FFV;
    const float* l2w  = lin2_w   + (size_t)l * DV * FFV;
    const float* l2b  = lin2_b   + (size_t)l * DV;
    const float* g0 = ln_g + ((size_t)l * 3 + 0) * DV;
    const float* g1 = ln_g + ((size_t)l * 3 + 1) * DV;
    const float* g2 = ln_g + ((size_t)l * 3 + 2) * DV;
    const float* b0 = ln_b + ((size_t)l * 3 + 0) * DV;
    const float* b1 = ln_b + ((size_t)l * 3 + 1) * DV;
    const float* b2 = ln_b + ((size_t)l * 3 + 2) * DV;

    // ---- self attention ----
    gemm_wmma<<<dim3(MROWS / 128, 3 * DV / 64), blk256, 0, stream>>>(
        h, saw, sab, t0, MROWS, 3 * DV, DV, 3 * DV, 0);
    attn_wmma<<<attn_grid, blk32, 0, stream>>>(t0, t1, 1);
    gemm_wmma<<<dim3(MROWS / 128, DV / 64), blk256, 0, stream>>>(
        t1, saow, saob, t2, MROWS, DV, DV, DV, 0);
    add_ln<<<MROWS, blk256, 0, stream>>>(h, t2, g0, b0);

    // ---- cross attention (q from h, k/v from x) ----
    gemm_wmma<<<dim3(MROWS / 128, DV / 64), blk256, 0, stream>>>(
        h, caw, cab, t0, MROWS, DV, DV, 3 * DV, 0);
    gemm_wmma<<<dim3(MROWS / 128, 2 * DV / 64), blk256, 0, stream>>>(
        x, caw + (size_t)DV * DV, cab + DV, t0 + DV, MROWS, 2 * DV, DV, 3 * DV, 0);
    attn_wmma<<<attn_grid, blk32, 0, stream>>>(t0, t1, 0);
    gemm_wmma<<<dim3(MROWS / 128, DV / 64), blk256, 0, stream>>>(
        t1, caow, caob, t2, MROWS, DV, DV, DV, 0);
    add_ln<<<MROWS, blk256, 0, stream>>>(h, t2, g1, b1);

    // ---- FFN ----
    gemm_wmma<<<dim3(MROWS / 128, FFV / 64), blk256, 0, stream>>>(
        h, l1w, l1b, t2, MROWS, FFV, DV, FFV, 1);
    gemm_wmma<<<dim3(MROWS / 128, DV / 64), blk256, 0, stream>>>(
        t2, l2w, l2b, t1, MROWS, DV, FFV, DV, 0);
    add_ln<<<MROWS, blk256, 0, stream>>>(h, t1, g2, b2);
  }

  cls_head<<<MROWS / 8, blk256, 0, stream>>>(h, cls_w, cls_b, out);
}
